// Model_81363860455701
// MI455X (gfx1250) — matrix operand order repaired and checked by reading, not compiled, not run on hardware
//
#include <hip/hip_runtime.h>

typedef __attribute__((ext_vector_type(16))) _Float16 v16h;
typedef __attribute__((ext_vector_type(8)))  _Float16 v8h;
typedef __attribute__((ext_vector_type(8)))  float    v8f;

union F16Frag { v16h v; v8h h[2]; };

#define NJ      24
#define ND      64
#define SWG     4            // samples per workgroup
#define ROWS    96           // SWG * NJ
#define STR     72           // padded LDS row stride in halves (144B, 16B aligned, bank-conflict free)
#define THREADS 96           // 3 waves, 2 row-tiles each
#define NBLK    256          // 1024 / SWG

__constant__ int PAR[24] = {-1,0,0,0,1,2,3,4,5,6,7,8,9,9,9,12,13,14,16,17,18,19,20,21};

__device__ __forceinline__ float sigm(float x){ return 1.0f/(1.0f+__expf(-x)); }

__device__ __forceinline__ v8f vz(){
  v8f z;
#pragma unroll
  for (int i=0;i<8;i++) z[i]=0.0f;
  return z;
}

// A fragment (16x32 f16): lane (hi*16+m) holds row m, halves K = k0+hi*8+[0..7] and +16+[0..7]
__device__ __forceinline__ v16h loadA(const _Float16* buf, int row, int k0, int hi){
  const _Float16* p = buf + row*STR + k0 + hi*8;
  F16Frag f; f.h[0] = *(const v8h*)p; f.h[1] = *(const v8h*)(p+16); return f.v;
}
// B fragment (32x16 f16) from transposed weight Wt[col][k] (stride 64): lane (hi*16+n) holds col n,
// halves K = k0+hi*8+[0..7] and +16+[0..7]
__device__ __forceinline__ v16h loadB(const _Float16* wt, int col, int k0, int hi){
  const _Float16* p = wt + col*64 + k0 + hi*8;
  F16Frag f; f.h[0] = *(const v8h*)p; f.h[1] = *(const v8h*)(p+16); return f.v;
}
// B fragment from transposed decoder weight (stride 1536)
__device__ __forceinline__ v16h loadBdec(const _Float16* wt, int col, int k0, int hi){
  const _Float16* p = wt + (size_t)col*1536 + k0 + hi*8;
  F16Frag f; f.h[0] = *(const v8h*)p; f.h[1] = *(const v8h*)(p+16); return f.v;
}

// One gate: g = agg_x@Wl[a] + x@Wr[a] + agg_h@Wl[b] + h@Wr[b]   (biases added by caller)
// wsW layout: 16 matrices [m][n][k] f16, m 0..7 = Wl, 8..15 = Wr
// Minimal-liveness schedule: one B frag + one A frag live at a time.
__device__ __forceinline__ void gate_gemm(const _Float16* __restrict__ wsW,
    int m0,int m1,int m2,int m3,
    const _Float16* sAX, const _Float16* sXb, const _Float16* sAH, const _Float16* sHb,
    int wave, int n, int hi, v8f g[2][4])
{
  const int mats[4] = {m0,m1,m2,m3};
  const _Float16* bufs[4] = {sAX,sXb,sAH,sHb};
#pragma unroll
  for (int rt=0;rt<2;rt++)
#pragma unroll
    for (int ct=0;ct<4;ct++) g[rt][ct]=vz();
#pragma unroll
  for (int ct=0;ct<4;ct++){
#pragma unroll
    for (int i=0;i<4;i++){
#pragma unroll
      for (int k=0;k<2;k++){
        v16h b = loadB(wsW + mats[i]*4096, ct*16+n, k*32, hi);
#pragma unroll
        for (int rt=0;rt<2;rt++){
          v16h a = loadA(bufs[i], wave*32 + rt*16 + n, k*32, hi);
          g[rt][ct] = __builtin_amdgcn_wmma_f32_16x16x32_f16(false, a, false, b,
                                                             (short)0, g[rt][ct], false, false);
        }
      }
    }
  }
}

// Prep: convert sage weights to f16 transposed [m][n][k], dec_W to f16 transposed [80][1536]
__global__ void prep_kernel(const float* __restrict__ Wl, const float* __restrict__ Wr,
                            const float* __restrict__ decW,
                            _Float16* __restrict__ wsW, _Float16* __restrict__ decWt)
{
  const int TOT = 16*4096 + 80*1536;
  for (int idx = blockIdx.x*blockDim.x + threadIdx.x; idx < TOT; idx += gridDim.x*blockDim.x){
    if (idx < 16*4096){
      int m = idx >> 12, rem = idx & 4095, nn = rem >> 6, kk = rem & 63;
      float v = (m < 8) ? Wl[m*4096 + kk*64 + nn] : Wr[(m-8)*4096 + kk*64 + nn];
      wsW[idx] = (_Float16)v;
    } else {
      int q = idx - 16*4096; int nn = q / 1536; int kk = q - nn*1536;
      decWt[q] = (_Float16)((nn < 72) ? decW[kk*72 + nn] : 0.0f);
    }
  }
}

__global__ __launch_bounds__(THREADS) void smpl_gclstm_kernel(
    const float* __restrict__ src, const float* __restrict__ tgt,
    const float* __restrict__ encW, const float* __restrict__ encb,
    const float* __restrict__ bl_in,  const float* __restrict__ gw_in,
    const float* __restrict__ gb_in,  const float* __restrict__ decb,
    const _Float16* __restrict__ wsW_in, const _Float16* __restrict__ decWt_in,
    float* __restrict__ dout)
{
  __shared__ _Float16 sX[ROWS*STR], sAX[ROWS*STR], sH[ROWS*STR], sAH[ROWS*STR];
  __shared__ int   nbrI[24][6];
  __shared__ int   nbrC[24];
  __shared__ float nbrS[24];

  const int tid  = threadIdx.x;
  const int wave = tid >> 5, lane = tid & 31;
  const int n = lane & 15, hi = lane >> 4;
  const int sbase = blockIdx.x * SWG;

  // build sparse normalized adjacency (self + parent + children, mean)
  if (tid < 24){
    int c = 0; nbrI[tid][c++] = tid;
    int p = PAR[tid]; if (p >= 0) nbrI[tid][c++] = p;
    for (int q = 0; q < 24; q++) if (PAR[q] == tid) nbrI[tid][c++] = q;
    for (int q = c; q < 6; q++) nbrI[tid][q] = tid;
    nbrC[tid] = c; nbrS[tid] = 1.0f/(float)c;
  }
  for (int i = tid; i < ROWS*STR; i += THREADS){
    sX[i]=(_Float16)0.0f; sAX[i]=(_Float16)0.0f; sH[i]=(_Float16)0.0f; sAH[i]=(_Float16)0.0f;
  }
  __syncthreads();

  // enc(src frame T_src-12 = 4) -> sX
  for (int i = tid; i < ROWS*ND; i += THREADS){
    int row = i >> 6, d = i & 63, s = row/24, j = row - s*24;
    const float* p = src + ((size_t)(sbase+s)*16 + 4)*72 + j*3;
    float v = encb[d] + p[0]*encW[d] + p[1]*encW[64+d] + p[2]*encW[128+d];
    sX[row*STR + d] = (_Float16)(v > 0.0f ? v : 0.0f);
  }
  __syncthreads();
  // agg(sX) -> sAX  (sAH stays 0, since h=0)
  for (int i = tid; i < ROWS*ND; i += THREADS){
    int row = i >> 6, d = i & 63, s = row/24, j = row - s*24;
    float sum = 0.0f; int c = nbrC[j];
    for (int q = 0; q < c; q++) sum += (float)sX[(s*24 + nbrI[j][q])*STR + d];
    sAX[row*STR + d] = (_Float16)(sum * nbrS[j]);
  }
  __syncthreads();

  v8f C[2][4], P[2][4];
#pragma unroll
  for (int rt=0;rt<2;rt++)
#pragma unroll
    for (int ct=0;ct<4;ct++) C[rt][ct]=vz();

#pragma unroll 1
  for (int t = 0; t < 24; t++){
    // Defeat LICM without destroying pointer provenance: launder a zero offset
    // through an SGPR so the weight/bias loads look loop-variant (no hoist+spill)
    // but still lower to global_load (saddr) instead of flat_load.
    long zoff = 0;
    asm volatile("" : "+s"(zoff));
    const _Float16* wsW   = wsW_in   + zoff;
    const _Float16* decWt = decWt_in + zoff;
    const float* bl = bl_in + zoff;
    const float* gw = gw_in + zoff;
    const float* gb = gb_in + zoff;

    v8f g[2][4];

    // I = sigmoid(sage(0,x)+sage(1,h) + gw0*c + gb0)   -> P (temporarily holds I)
    gate_gemm(wsW, 0, 8, 1, 9, sAX, sX, sAH, sH, wave, n, hi, g);
#pragma unroll
    for (int rt=0;rt<2;rt++)
#pragma unroll
      for (int ct=0;ct<4;ct++){
        int col = ct*16 + n;
        float bs = bl[col] + bl[64+col], w0 = gw[col], b0 = gb[col];
#pragma unroll
        for (int r=0;r<8;r++)
          P[rt][ct][r] = sigm(g[rt][ct][r] + bs + w0*C[rt][ct][r] + b0);
      }

    // T = tanh(sage(4,x)+sage(5,h) + gb2);  P = I*T  (I dies here)
    gate_gemm(wsW, 4, 12, 5, 13, sAX, sX, sAH, sH, wave, n, hi, g);
#pragma unroll
    for (int rt=0;rt<2;rt++)
#pragma unroll
      for (int ct=0;ct<4;ct++){
        int col = ct*16 + n;
        float bs = bl[256+col] + bl[320+col], b2 = gb[128+col];
#pragma unroll
        for (int r=0;r<8;r++)
          P[rt][ct][r] *= tanhf(g[rt][ct][r] + bs + b2);
      }

    // F = sigmoid(sage(2,x)+sage(3,h) + gw1*c + gb1);  c = F*c + P  (P dies)
    gate_gemm(wsW, 2, 10, 3, 11, sAX, sX, sAH, sH, wave, n, hi, g);
#pragma unroll
    for (int rt=0;rt<2;rt++)
#pragma unroll
      for (int ct=0;ct<4;ct++){
        int col = ct*16 + n;
        float bs = bl[128+col] + bl[192+col], w1 = gw[64+col], b1 = gb[64+col];
#pragma unroll
        for (int r=0;r<8;r++){
          float f = sigm(g[rt][ct][r] + bs + w1*C[rt][ct][r] + b1);
          C[rt][ct][r] = f*C[rt][ct][r] + P[rt][ct][r];
        }
      }

    // O = sigmoid(sage(6,x)+sage(7,h) + gw2*c_new + gb3)   (h recomputed at write)
    v8f Og[2][4];
    gate_gemm(wsW, 6, 14, 7, 15, sAX, sX, sAH, sH, wave, n, hi, g);
#pragma unroll
    for (int rt=0;rt<2;rt++)
#pragma unroll
      for (int ct=0;ct<4;ct++){
        int col = ct*16 + n;
        float bs = bl[384+col] + bl[448+col], w2 = gw[128+col], b3 = gb[192+col];
#pragma unroll
        for (int r=0;r<8;r++)
          Og[rt][ct][r] = sigm(g[rt][ct][r] + bs + w2*C[rt][ct][r] + b3);
      }

    __syncthreads();   // all GEMM reads of sX/sAX/sH/sAH complete

    // write next-step x (=O, except t==11 which re-encodes) and h = O*tanh(c)
#pragma unroll
    for (int rt=0;rt<2;rt++)
#pragma unroll
      for (int ct=0;ct<4;ct++){
        int col = ct*16 + n;
#pragma unroll
        for (int r=0;r<8;r++){
          int row = wave*32 + rt*16 + r + (hi ? 8 : 0);
          float o = Og[rt][ct][r];
          if (t != 11) sX[row*STR + col] = (_Float16)o;
          sH[row*STR + col] = (_Float16)(o * tanhf(C[rt][ct][r]));
        }
      }
    if (t == 11){
      for (int i = tid; i < ROWS*ND; i += THREADS){
        int row = i >> 6, d = i & 63, s = row/24, j = row - s*24;
        const float* p = tgt + ((size_t)(sbase+s)*12)*72 + j*3;
        float v = encb[d] + p[0]*encW[d] + p[1]*encW[64+d] + p[2]*encW[128+d];
        sX[row*STR + d] = (_Float16)(v > 0.0f ? v : 0.0f);
      }
    }
    __syncthreads();

    // decode output: out[:, t-12, :] = O_flat[S,1536] @ dec_W + dec_b, via WMMA vs decWt[80][1536]
    if (t >= 12){
      const int nct = (wave < 2) ? 2 : 1, ct0 = wave*2;
      v8f dacc[2]; dacc[0]=vz(); dacc[1]=vz();
#pragma unroll 2
      for (int kb = 0; kb < 1536; kb += 32){
        int k1 = kb + hi*8, k2 = k1 + 16;
        const _Float16* pa1 = sX + ((n & 3)*24 + (k1 >> 6))*STR + (k1 & 63);
        const _Float16* pa2 = sX + ((n & 3)*24 + (k2 >> 6))*STR + (k2 & 63);
        F16Frag fa; fa.h[0] = *(const v8h*)pa1; fa.h[1] = *(const v8h*)pa2;
        for (int c = 0; c < nct; c++){
          v16h b = loadBdec(decWt, (ct0+c)*16 + n, kb, hi);
          dacc[c] = __builtin_amdgcn_wmma_f32_16x16x32_f16(false, fa.v, false, b,
                                                           (short)0, dacc[c], false, false);
        }
      }
      for (int c = 0; c < nct; c++){
        int col = (ct0+c)*16 + n;
        if (hi == 0 && col < 72){
          float bias = decb[col];
#pragma unroll
          for (int r = 0; r < SWG; r++)
            dout[((size_t)(sbase+r)*12 + (t-12))*72 + col] = dacc[c][r] + bias;
        }
      }
    }

    // adjacency aggregation for next step
    if (t < 23){
      for (int i = tid; i < ROWS*ND; i += THREADS){
        int row = i >> 6, d = i & 63, s = row/24, j = row - s*24;
        float sx = 0.0f, sh = 0.0f; int c = nbrC[j];
        for (int q = 0; q < c; q++){
          int rr = (s*24 + nbrI[j][q])*STR + d;
          sx += (float)sX[rr];
          sh += (float)sH[rr];
        }
        sAX[row*STR + d] = (_Float16)(sx * nbrS[j]);
        sAH[row*STR + d] = (_Float16)(sh * nbrS[j]);
      }
      __syncthreads();
    }
  }
}

extern "C" void kernel_launch(void* const* d_in, const int* in_sizes, int n_in,
                              void* d_out, int out_size, void* d_ws, size_t ws_size,
                              hipStream_t stream) {
  (void)in_sizes; (void)n_in; (void)out_size; (void)ws_size;
  const float* src  = (const float*)d_in[0];
  const float* tgt  = (const float*)d_in[1];
  const float* encW = (const float*)d_in[2];
  const float* encb = (const float*)d_in[3];
  const float* Wl   = (const float*)d_in[4];
  const float* bl   = (const float*)d_in[5];
  const float* Wr   = (const float*)d_in[6];
  const float* gw   = (const float*)d_in[7];
  const float* gb   = (const float*)d_in[8];
  const float* decW = (const float*)d_in[9];
  const float* decb = (const float*)d_in[10];

  _Float16* wsW   = (_Float16*)d_ws;        // 16*64*64 halves = 128KB
  _Float16* decWt = wsW + 16*4096;          // 80*1536 halves  = 240KB

  prep_kernel<<<736, 256, 0, stream>>>(Wl, Wr, decW, wsW, decWt);
  smpl_gclstm_kernel<<<NBLK, THREADS, 0, stream>>>(src, tgt, encW, encb, bl, gw, gb, decb,
                                                   wsW, decWt, (float*)d_out);
}
